// ADConv_38551626449495
// MI455X (gfx1250) — matrix operand order repaired and checked by reading, not compiled, not run on hardware
//
#include <hip/hip_runtime.h>
#include <hip/hip_bf16.h>

typedef __attribute__((ext_vector_type(16))) __bf16  v16bf;
typedef __attribute__((ext_vector_type(8)))  float   v8f;

#define N_     16
#define C_     64
#define H_     64
#define W_     64
#define INTER_ 64
#define BCH_   96     // TOTAL_BASES * NUM_FA
#define O_     128
#define HP_    66     // H + 2 (pad=1)
#define FA_    16
#define TB_    6
#define DTOT_  1024   // C_ * FA_
#define NKS_CONV 18   // (C_*9)/32
#define NKS_FIN  32   // DTOT_/32
#define CHW_   ((long)HP_ * HP_)   // 4356 elements per (n,c) plane

static __device__ __forceinline__ unsigned short f2bf(float f) {
    unsigned u = __float_as_uint(f);
    u += 0x7FFFu + ((u >> 16) & 1u);          // round-to-nearest-even
    return (unsigned short)(u >> 16);
}
static __device__ __forceinline__ float bf2f(unsigned short h) {
    return __uint_as_float(((unsigned)h) << 16);
}

// ---------------------------------------------------------------------------
// Prep: build padded bf16 x, zero padded h buffer, pack weights into the
// per-lane WMMA B-fragment linear layout (lane holds N=lane&15, K=(j&7)+((j&8)<<1)+((lane&16)>>1)).
// Conv K-ordering: K = r*64 + c  (r = kh*3+kw tap index, c = input channel),
// so each 32-wide k-step has a constant tap and compile-time channel offsets.
// ---------------------------------------------------------------------------
__global__ void adconv_prep(const float* __restrict__ x,
                            const float* __restrict__ w1,
                            const float* __restrict__ w2,
                            const float* __restrict__ coef,
                            unsigned short* __restrict__ xpad,
                            unsigned short* __restrict__ hpad,
                            unsigned short* __restrict__ w1pk,
                            unsigned short* __restrict__ w2pk,
                            unsigned short* __restrict__ cfpk)
{
    long i = (long)blockIdx.x * blockDim.x + threadIdx.x;
    const long XP  = (long)N_ * C_ * HP_ * HP_;
    const long HPD = (long)N_ * INTER_ * HP_ * HP_;
    const long W1  = 4L * NKS_CONV * 32 * 16;     // 36864
    const long W2  = 6L * NKS_CONV * 32 * 16;     // 55296
    const long CF  = 8L * NKS_FIN  * 32 * 16;     // 131072

    if (i < XP) {
        int ww = (int)(i % HP_); long t = i / HP_;
        int hh = (int)(t % HP_); t /= HP_;
        int c  = (int)(t % C_);  int n = (int)(t / C_);
        unsigned short v = 0;
        if (hh >= 1 && hh <= H_ && ww >= 1 && ww <= W_)
            v = f2bf(x[(((long)n * C_ + c) * H_ + (hh - 1)) * W_ + (ww - 1)]);
        xpad[i] = v;
        return;
    }
    i -= XP;
    if (i < HPD) { hpad[i] = 0; return; }
    i -= HPD;
    if (i < W1) {
        int ii = (int)i;
        int j = ii & 15, lane = (ii >> 4) & 31;
        int ks = (ii >> 9) % NKS_CONV, nt = (ii >> 9) / NKS_CONV;
        int o = nt * 16 + (lane & 15);
        int K = ks * 32 + (j & 7) + ((j & 8) << 1) + ((lane & 16) >> 1);
        int c = K & 63, r = K >> 6;               // K = r*64 + c
        w1pk[ii] = f2bf(w1[(o * C_ + c) * 9 + r]);
        return;
    }
    i -= W1;
    if (i < W2) {
        int ii = (int)i;
        int j = ii & 15, lane = (ii >> 4) & 31;
        int ks = (ii >> 9) % NKS_CONV, nt = (ii >> 9) / NKS_CONV;
        int o = nt * 16 + (lane & 15);
        int K = ks * 32 + (j & 7) + ((j & 8) << 1) + ((lane & 16) >> 1);
        int c = K & 63, r = K >> 6;
        w2pk[ii] = f2bf(w2[(o * C_ + c) * 9 + r]);
        return;
    }
    i -= W2;
    if (i < CF) {
        int ii = (int)i;
        int j = ii & 15, lane = (ii >> 4) & 31;
        int ks = (ii >> 9) & 31, nt = ii >> 14;
        int o = nt * 16 + (lane & 15);
        int d = ks * 32 + (j & 7) + ((j & 8) << 1) + ((lane & 16) >> 1);
        cfpk[ii] = f2bf(coef[(long)o * DTOT_ + d]);
    }
}

// ---------------------------------------------------------------------------
// Implicit-GEMM 3x3 conv + bias + BN + tanh via v_wmma_f32_16x16x32_bf16.
// One wave = 16 pixels (consecutive w) x 16 output channels; K = 9*64 = 576
// ordered K = tap*64 + channel: per k-step the tap is fixed, so the 16 A-lane
// elements are base + compile-time-constant * plane-stride (immediate IOFFSET).
// mode 0: write bf16 into padded dst (feeds next conv). mode 1: write f32.
// ---------------------------------------------------------------------------
__global__ void adconv_conv_wmma(const unsigned short* __restrict__ src,  // (N,Cin,66,66) bf16
                                 const unsigned short* __restrict__ wpk,
                                 const float* __restrict__ bias,
                                 const float* __restrict__ gam,
                                 const float* __restrict__ bet,
                                 const float* __restrict__ mu,
                                 const float* __restrict__ var,
                                 unsigned short* __restrict__ dst_pad,
                                 float* __restrict__ dst_f32,
                                 int Cin, int Cout, int mode)
{
    const int lane = threadIdx.x & 31;
    const int nt   = threadIdx.x >> 5;                 // output-channel tile
    const int tile = blockIdx.x;
    const int w0   = (tile & 3) << 4;
    const int h    = (tile >> 2) & 63;
    const int n    = tile >> 8;
    const int M    = lane & 15;                        // A-matrix row (pixel)

    // Fold lane-half K offset (+8 channels) into the base pointer once.
    const unsigned short* srcB = src + (long)n * Cin * CHW_
                               + ((long)((lane & 16) >> 1)) * CHW_
                               + (w0 + M);

    v8f acc = {};
    for (int ks = 0; ks < NKS_CONV; ++ks) {
        const int r  = ks >> 1;                        // tap index kh*3+kw
        const int c0 = (ks & 1) << 5;                  // channel base 0 / 32
        const int kh = r / 3, kw = r - 3 * kh;
        const unsigned short* base = srcB + (long)c0 * CHW_
                                   + (long)(h + kh) * HP_ + kw;
        union { v16bf v; unsigned short u[16]; } A;
        #pragma unroll
        for (int j = 0; j < 16; ++j) {
            const int Kl = (j & 7) | ((j & 8) << 1);   // compile-time constant
            A.u[j] = base[(long)Kl * CHW_];
        }
        union { v16bf v; uint4 q[2]; } B;
        const uint4* pb = (const uint4*)(wpk + (((long)(nt * NKS_CONV + ks) * 32 + lane) << 4));
        B.q[0] = pb[0]; B.q[1] = pb[1];
        if (ks + 1 < NKS_CONV)
            __builtin_prefetch((const void*)(pb + 64), 0, 0);  // next k-step B line
        acc = __builtin_amdgcn_wmma_f32_16x16x32_bf16(false, A.v, false, B.v,
                                                      (short)0, acc, false, false);
    }

    const int ch = nt * 16 + (lane & 15);              // D-matrix column = out channel
    const float sc = gam[ch] * rsqrtf(var[ch] + 1e-5f);
    const float b0 = bias[ch] - mu[ch];
    const float bt = bet[ch];
    const int Mb = (lane & 16) >> 1;                   // D rows 0..7 / 8..15
    #pragma unroll
    for (int e = 0; e < 8; ++e) {
        int m = Mb + e;
        float v = tanhf((acc[e] + b0) * sc + bt);
        if (mode == 0)
            dst_pad[(((long)n * Cout + ch) * HP_ + (h + 1)) * HP_ + (1 + w0 + m)] = f2bf(v);
        else
            dst_f32[(((long)n * Cout + ch) * H_ + h) * (long)W_ + (w0 + m)] = v;
    }
}

// ---------------------------------------------------------------------------
// Fused atoms + bases_out + final 1x1 GEMM.
// Block = 256 threads = 8 waves; 16 pixels (consecutive a) of row b.
// Phase 1: build bases_out tile (16 x 1024, bf16) in LDS.
// Phase 2: 8 waves x WMMA over K=1024 against packed coef; store out[n,o,b,a].
// ---------------------------------------------------------------------------
__global__ void adconv_final_wmma(const unsigned short* __restrict__ xpad,
                                  const float* __restrict__ bc,     // (N,96,H,W) f32
                                  const float* __restrict__ bases,  // (6,9) f32
                                  const unsigned short* __restrict__ cfpk,
                                  float* __restrict__ out)          // (N,O,W,H)
{
    __shared__ float sBases[TB_ * 9];
    __shared__ unsigned short sA[16 * DTOT_];          // 32 KB
    const int t = threadIdx.x;
    if (t < TB_ * 9) sBases[t] = bases[t];
    __syncthreads();

    const int a0 = (blockIdx.x & 3) << 4;
    const int b  = (blockIdx.x >> 2) & 63;
    const int n  = blockIdx.x >> 8;

    // ---- phase 1: atoms + bases_out -> LDS ----
    {
        const int p = t >> 4, f = t & 15;
        const int a = a0 + p;
        float atoms[9];
        #pragma unroll
        for (int k = 0; k < 9; ++k) atoms[k] = 0.f;
        #pragma unroll
        for (int tt = 0; tt < TB_; ++tt) {
            // reference transpose(0,3,2,1): bases_coef[n,a,b,f,tt] = bc[n, f*6+tt, b, a]
            float cv = bc[(((long)n * BCH_ + (f * TB_ + tt)) * H_ + b) * (long)W_ + a];
            #pragma unroll
            for (int k = 0; k < 9; ++k) atoms[k] += cv * sBases[tt * 9 + k];
        }
        const unsigned short* px0 = xpad + ((long)n * C_ * CHW_) + (long)a * HP_ + b;
        for (int c = 0; c < C_; ++c) {
            const unsigned short* px = px0 + (long)c * CHW_;
            float s = 0.f;
            #pragma unroll
            for (int kh = 0; kh < 3; ++kh)
                #pragma unroll
                for (int kw = 0; kw < 3; ++kw)
                    s += bf2f(px[kh * HP_ + kw]) * atoms[kh * 3 + kw];
            sA[p * DTOT_ + c * FA_ + f] = f2bf(s);     // d = c*16 + f (C-major flatten)
        }
    }
    __syncthreads();

    // ---- phase 2: (16 pixels) x (16 out-ch per wave), K = 1024 ----
    const int lane = t & 31, ot = t >> 5;
    const int kd   = (lane & 16) >> 1;
    const int mrow = lane & 15;
    v8f acc = {};
    for (int ks = 0; ks < NKS_FIN; ++ks) {
        union { v16bf v; unsigned u32[8]; } A;
        #pragma unroll
        for (int jj = 0; jj < 8; ++jj) {
            int Kl = ((2 * jj) & 7) | (((2 * jj) & 8) << 1) | kd;   // even -> 4B aligned
            A.u32[jj] = *(const unsigned*)&sA[mrow * DTOT_ + ks * 32 + Kl];
        }
        union { v16bf v; uint4 q[2]; } B;
        const uint4* pb = (const uint4*)(cfpk + (((long)(ot * NKS_FIN + ks) * 32 + lane) << 4));
        B.q[0] = pb[0]; B.q[1] = pb[1];
        if (ks + 1 < NKS_FIN)
            __builtin_prefetch((const void*)(pb + 64), 0, 0);  // next k-step B line
        acc = __builtin_amdgcn_wmma_f32_16x16x32_bf16(false, A.v, false, B.v,
                                                      (short)0, acc, false, false);
    }
    const int o  = ot * 16 + (lane & 15);
    const int Mb = (lane & 16) >> 1;
    #pragma unroll
    for (int e = 0; e < 8; ++e)
        out[(((long)n * O_ + o) * W_ + b) * (long)H_ + (a0 + Mb + e)] = acc[e];
}

// ---------------------------------------------------------------------------
extern "C" void kernel_launch(void* const* d_in, const int* in_sizes, int n_in,
                              void* d_out, int out_size, void* d_ws, size_t ws_size,
                              hipStream_t stream)
{
    (void)in_sizes; (void)n_in; (void)out_size; (void)ws_size;
    const float* x    = (const float*)d_in[0];
    const float* w1   = (const float*)d_in[1];
    const float* b1   = (const float*)d_in[2];
    const float* g1   = (const float*)d_in[3];
    const float* be1  = (const float*)d_in[4];
    const float* mu1  = (const float*)d_in[5];
    const float* var1 = (const float*)d_in[6];
    const float* w2   = (const float*)d_in[7];
    const float* b2   = (const float*)d_in[8];
    const float* g2   = (const float*)d_in[9];
    const float* be2  = (const float*)d_in[10];
    const float* mu2  = (const float*)d_in[11];
    const float* var2 = (const float*)d_in[12];
    const float* bases = (const float*)d_in[13];
    const float* coef  = (const float*)d_in[14];
    float* out = (float*)d_out;

    char* ws = (char*)d_ws;
    unsigned short* xpad = (unsigned short*)(ws + 0);            //  8,921,088 B
    unsigned short* hpad = (unsigned short*)(ws + 8921088);      //  8,921,088 B
    float*          bcb  = (float*)         (ws + 17842176);     // 25,165,824 B
    unsigned short* w1pk = (unsigned short*)(ws + 43008000);     //     73,728 B
    unsigned short* w2pk = (unsigned short*)(ws + 43081728);     //    110,592 B
    unsigned short* cfpk = (unsigned short*)(ws + 43192320);     //    262,144 B

    const long totalPrep = (long)N_ * C_ * HP_ * HP_        // xpad
                         + (long)N_ * INTER_ * HP_ * HP_    // hpad zero
                         + 4L * NKS_CONV * 32 * 16
                         + 6L * NKS_CONV * 32 * 16
                         + 8L * NKS_FIN  * 32 * 16;
    const int prepBlocks = (int)((totalPrep + 255) / 256);

    adconv_prep<<<prepBlocks, 256, 0, stream>>>(x, w1, w2, coef,
                                                xpad, hpad, w1pk, w2pk, cfpk);

    // conv1: 64 out-ch -> 4 waves/block
    adconv_conv_wmma<<<4096, 128, 0, stream>>>(xpad, w1pk, b1, g1, be1, mu1, var1,
                                               hpad, nullptr, C_, INTER_, 0);
    // conv2: 96 out-ch -> 6 waves/block
    adconv_conv_wmma<<<4096, 192, 0, stream>>>(hpad, w2pk, b2, g2, be2, mu2, var2,
                                               nullptr, bcb, INTER_, BCH_, 1);
    // fused atoms + bases_out + final GEMM
    adconv_final_wmma<<<4096, 256, 0, stream>>>(xpad, bcb, bases, cfpk, out);
}
